// TCN_GCN_unit_18176301597609
// MI455X (gfx1250) — matrix operand order repaired and checked by reading, not compiled, not run on hardware
//
#include <hip/hip_runtime.h>
#include <hip/hip_bf16.h>

// ---------------- problem constants ----------------
constexpr int kN  = 128;
constexpr int kT  = 128;
constexpr int kV  = 25;
constexpr int kP  = kN * kT * kV;   // 409600 positions
constexpr int kNT = kN * kT;        // 16384
constexpr float kEPS = 1e-5f;

typedef __attribute__((ext_vector_type(16))) __bf16 v16bf;
typedef __attribute__((ext_vector_type(8)))  __bf16 v8bf;
typedef __attribute__((ext_vector_type(2)))  __bf16 v2bf;
typedef __attribute__((ext_vector_type(8)))  float  v8f;
typedef __attribute__((ext_vector_type(4)))  float  v4f;
typedef __attribute__((ext_vector_type(2)))  float  v2f;

__device__ __forceinline__ v8f wmma_bf16(v16bf a, v16bf b, v8f c) {
  return __builtin_amdgcn_wmma_f32_16x16x32_bf16(false, a, false, b, (short)0, c,
                                                 false, false);
}

__device__ __forceinline__ void decode_p(int p, int& n, int& t, int& v) {
  n = p / (kT * kV);
  int r = p - n * (kT * kV);
  t = r / kV;
  v = r - t * kV;
}

// ---------------- prep: zero stats, convert weights to bf16 ----------------
__global__ __launch_bounds__(256) void k_prep(const float* __restrict__ gdw,
                                              const float* __restrict__ gsw,
                                              const float* __restrict__ brw,
                                              __bf16* __restrict__ gdwb,
                                              __bf16* __restrict__ gswb,
                                              __bf16* __restrict__ brwb,
                                              float* __restrict__ stats) {
  int i = blockIdx.x * 256 + threadIdx.x;
  if (i < 2048) stats[i] = 0.f;
  if (i < 3 * 16 * 64) gdwb[i] = (__bf16)gdw[i];
  if (i < 4 * 16 * 64) brwb[i] = (__bf16)brw[i];
  if (i < 15 * 16 * 32) {                 // K padded 16 -> 32 with zeros
    int k = i & 31;
    int row = i >> 5;                     // (l*5+j)*16 + cout
    gswb[i] = (k < 16) ? (__bf16)gsw[row * 16 + k] : (__bf16)0.f;
  }
}

__global__ __launch_bounds__(256) void k_zero_f32(float* __restrict__ p, int n) {
  for (int i = blockIdx.x * 256 + threadIdx.x; i < n; i += gridDim.x * 256)
    p[i] = 0.f;
}

// ---------------- pack x (N,C,T,V) f32 -> (P,64) bf16 ----------------
__global__ __launch_bounds__(256) void k_pack_x(const float* __restrict__ x,
                                                __bf16* __restrict__ xb) {
  for (int i = blockIdx.x * 256 + threadIdx.x; i < kP * 8; i += gridDim.x * 256) {
    int c0 = (i & 7) * 8, p = i >> 3;
    int n, t, v; decode_p(p, n, t, v);
    v8bf o;
#pragma unroll
    for (int e = 0; e < 8; ++e)
      o[e] = (__bf16)x[((n * 64 + c0 + e) * kT + t) * kV + v];
    *(v8bf*)(xb + (size_t)p * 64 + c0) = o;
  }
}

// ---- shared epilogue: store C tile via LDS repack -> global_store_b128 ----
__device__ __forceinline__ void store_tile_repack(
    __bf16* __restrict__ stile,          // 256 bf16 per-wave slice
    __bf16* __restrict__ Y, int tile, int lane, int row, int hi,
    const v8f& c, float bias_n, float& lsum, float& lsq) {
#pragma unroll
  for (int g = 0; g < 8; ++g) {
    float val = c[g] + bias_n;
    stile[(hi * 8 + g) * 16 + row] = (__bf16)val;
    lsum += val; lsq += val * val;
  }
  asm volatile("s_wait_dscnt 0" ::: "memory");
  const int r = lane >> 1, h2 = lane & 1;
  v8bf pk = *(const v8bf*)&stile[r * 16 + h2 * 8];
  *(v8bf*)(Y + (size_t)tile * 256 + r * 16 + h2 * 8) = pk;
}

// ---------------- WMMA GEMM: [P,64] x [16,64]^T -> [P,16] (+bias, BN stats) ----
__global__ __launch_bounds__(256) void k_gemm_nk64(const __bf16* __restrict__ A,
                                                   const __bf16* __restrict__ W,
                                                   const float* __restrict__ bias,
                                                   __bf16* __restrict__ Y,
                                                   float* __restrict__ stats) {
  __shared__ float ssum[16], ssq[16];
  __shared__ __bf16 stile[8][256];
  if (threadIdx.x < 16) { ssum[threadIdx.x] = 0.f; ssq[threadIdx.x] = 0.f; }
  __syncthreads();

  const int lane = threadIdx.x & 31;
  const int wv   = threadIdx.x >> 5;
  const int row  = lane & 15;
  const int hi   = lane >> 4;

  const v8bf* wp0 = (const v8bf*)(W + row * 64 + hi * 16);
  const v8bf* wp1 = (const v8bf*)(W + row * 64 + 32 + hi * 16);
  v8bf w00 = wp0[0], w01 = wp0[1], w10 = wp1[0], w11 = wp1[1];
  v16bf b0, b1;
#pragma unroll
  for (int e = 0; e < 8; ++e) {
    b0[e] = w00[e]; b0[e + 8] = w01[e];
    b1[e] = w10[e]; b1[e + 8] = w11[e];
  }
  const float bias_n = bias[row];

  float lsum = 0.f, lsq = 0.f;
  const int tiles = kP >> 4;
  const int gw = blockIdx.x * 8 + wv;
  const int nw = gridDim.x * 8;
  for (int tile = gw; tile < tiles; tile += nw) {
    const int p = tile * 16 + row;
    const v8bf* ap = (const v8bf*)(A + (size_t)p * 64);
    __builtin_prefetch(A + ((size_t)p + (size_t)nw * 16) * 64, 0, 0);
    v8bf a00 = ap[2 * hi], a01 = ap[2 * hi + 1], a10 = ap[4 + 2 * hi], a11 = ap[5 + 2 * hi];
    v16bf a0, a1;
#pragma unroll
    for (int e = 0; e < 8; ++e) {
      a0[e] = a00[e]; a0[e + 8] = a01[e];
      a1[e] = a10[e]; a1[e + 8] = a11[e];
    }
    v8f c = {};
    c = wmma_bf16(a0, b0, c);
    c = wmma_bf16(a1, b1, c);
    store_tile_repack(&stile[wv][0], Y, tile, lane, row, hi, c, bias_n, lsum, lsq);
  }
  atomicAdd(&ssum[row], lsum);
  atomicAdd(&ssq[row], lsq);
  __syncthreads();
  if (threadIdx.x < 16) {
    atomicAdd(&stats[threadIdx.x], ssum[threadIdx.x]);
    atomicAdd(&stats[16 + threadIdx.x], ssq[threadIdx.x]);
  }
}

// ---------------- WMMA GEMM: [P,16] x [16,32-padded]^T -> [P,16] ----------------
__global__ __launch_bounds__(256) void k_gemm_nk16(const __bf16* __restrict__ A,
                                                   const __bf16* __restrict__ Wp,
                                                   const float* __restrict__ bias,
                                                   __bf16* __restrict__ Y,
                                                   float* __restrict__ stats) {
  __shared__ float ssum[16], ssq[16];
  __shared__ __bf16 stile[8][256];
  if (threadIdx.x < 16) { ssum[threadIdx.x] = 0.f; ssq[threadIdx.x] = 0.f; }
  __syncthreads();

  const int lane = threadIdx.x & 31;
  const int wv   = threadIdx.x >> 5;
  const int row  = lane & 15;
  const int hi   = lane >> 4;

  const v8bf* wp = (const v8bf*)(Wp + row * 32 + hi * 8);  // w1 (k>=16) is zeros
  v8bf w0 = wp[0], w1 = wp[2];
  v16bf b;
#pragma unroll
  for (int e = 0; e < 8; ++e) { b[e] = w0[e]; b[e + 8] = w1[e]; }
  const float bias_n = bias[row];

  float lsum = 0.f, lsq = 0.f;
  const int tiles = kP >> 4;
  const int gw = blockIdx.x * 8 + wv;
  const int nw = gridDim.x * 8;
  for (int tile = gw; tile < tiles; tile += nw) {
    const int p = tile * 16 + row;
    const v8bf* ap = (const v8bf*)(A + (size_t)p * 16);
    __builtin_prefetch(A + ((size_t)p + (size_t)nw * 16) * 16, 0, 0);
    v8bf av = ap[hi];
    v16bf a;
#pragma unroll
    for (int e = 0; e < 8; ++e) { a[e] = av[e]; a[e + 8] = (__bf16)0.f; }
    v8f c = {};
    c = wmma_bf16(a, b, c);
    store_tile_repack(&stile[wv][0], Y, tile, lane, row, hi, c, bias_n, lsum, lsq);
  }
  atomicAdd(&ssum[row], lsum);
  atomicAdd(&ssq[row], lsq);
  __syncthreads();
  if (threadIdx.x < 16) {
    atomicAdd(&stats[threadIdx.x], ssum[threadIdx.x]);
    atomicAdd(&stats[16 + threadIdx.x], ssq[threadIdx.x]);
  }
}

// ---------------- BN finalize ----------------
__global__ void k_finalize16(float* __restrict__ slot,
                             const float* __restrict__ g,
                             const float* __restrict__ b, float invN) {
  int c = threadIdx.x;
  if (c >= 16) return;
  float mean = slot[c] * invN;
  float var  = slot[16 + c] * invN - mean * mean;
  float sc   = g[c] * rsqrtf(var + kEPS);
  slot[32 + c] = sc;
  slot[48 + c] = b[c] - mean * sc;
}

__global__ void k_finalize64(float* __restrict__ slot,
                             const float* __restrict__ g,
                             const float* __restrict__ b, float invN) {
  int c = threadIdx.x;
  if (c >= 64) return;
  float mean = slot[c] * invN;
  float var  = slot[64 + c] * invN - mean * mean;
  float sc   = g[c] * rsqrtf(var + kEPS);
  slot[128 + c] = sc;
  slot[192 + c] = b[c] - mean * sc;
}

// ---------------- apply BN (+ReLU) on [P,16] bf16, 8 elems/thread ----------------
__global__ __launch_bounds__(256) void k_apply16_relu(const __bf16* __restrict__ in,
                                                      const float* __restrict__ slot,
                                                      __bf16* __restrict__ out) {
  for (int i = blockIdx.x * 256 + threadIdx.x; i < kP * 2; i += gridDim.x * 256) {
    int mb = (i & 1) * 8;
    size_t off = (size_t)(i >> 1) * 16 + mb;
    v8bf a = *(const v8bf*)(in + off);
    v8bf o;
#pragma unroll
    for (int e = 0; e < 8; ++e) {
      float v = (float)a[e] * slot[32 + mb + e] + slot[48 + mb + e];
      o[e] = (__bf16)fmaxf(v, 0.f);
    }
    *(v8bf*)(out + off) = o;
  }
}

// ---- graph mix over V: zv[nt,v,c] = sum_u PA[v,u]*xd[nt,u,c]; float2 channel pair
__global__ __launch_bounds__(256) void k_vmix(const __bf16* __restrict__ xd,
                                              const float* __restrict__ PA,
                                              __bf16* __restrict__ zv) {
  __shared__ float sPA[kV * kV];
  // CDNA5 async copy global -> LDS, tracked by ASYNCcnt
  for (int i = threadIdx.x; i < kV * kV; i += 256) {
    unsigned lds = (unsigned)(size_t)(&sPA[i]);
    const float* gp = PA + i;
    asm volatile("global_load_async_to_lds_b32 %0, %1, off"
                 :: "v"(lds), "v"(gp) : "memory");
  }
  asm volatile("s_wait_asynccnt 0" ::: "memory");
  __syncthreads();

  const int g  = blockIdx.x * 256 + threadIdx.x;   // over NT*8 (exact grid)
  const int cp = g & 7, nt = g >> 3;
  const size_t base = (size_t)nt * kV * 16 + cp * 2;
  v2f in[kV];
#pragma unroll
  for (int u = 0; u < kV; ++u) {
    v2bf t = *(const v2bf*)(xd + base + u * 16);
    in[u] = v2f{(float)t[0], (float)t[1]};
  }
  for (int v = 0; v < kV; ++v) {
    v2f acc = {0.f, 0.f};
#pragma unroll
    for (int u = 0; u < kV; ++u) acc += in[u] * sPA[v * kV + u];
    v2bf o; o[0] = (__bf16)acc[0]; o[1] = (__bf16)acc[1];
    *(v2bf*)(zv + base + v * 16) = o;
  }
}

// ---------------- accumulate normalized ys into agg (f32 [P,64]) ----------------
__global__ __launch_bounds__(256) void k_accum_agg(const __bf16* __restrict__ ys,
                                                   const float* __restrict__ slot,
                                                   float* __restrict__ agg, int j) {
  for (int i = blockIdx.x * 256 + threadIdx.x; i < kP * 2; i += gridDim.x * 256) {
    int mb = (i & 1) * 8, p = i >> 1;
    v8bf y = *(const v8bf*)(ys + (size_t)p * 16 + mb);
    v4f c0, c1;
#pragma unroll
    for (int e = 0; e < 4; ++e) {
      c0[e] = (float)y[e]     * slot[32 + mb + e]     + slot[48 + mb + e];
      c1[e] = (float)y[e + 4] * slot[32 + mb + e + 4] + slot[48 + mb + e + 4];
    }
    float* ap = agg + (size_t)p * 64;
    if (j == 0) {
#pragma unroll
      for (int q = 0; q < 4; ++q) {
        v4f* d0 = (v4f*)(ap + q * 16 + mb);
        v4f* d1 = d0 + 1;
        *d0 += c0; *d1 += c1;
      }
    } else {
      v4f* d0 = (v4f*)(ap + (j - 1) * 16 + mb);
      v4f* d1 = d0 + 1;
      *d0 += c0; *d1 += c1;
    }
  }
}

// ---------------- per-channel stats of agg (64 ch) ----------------
__global__ __launch_bounds__(256) void k_stats64(const float* __restrict__ agg,
                                                 float* __restrict__ slot) {
  __shared__ float ssum[64], ssq[64];
  if (threadIdx.x < 64) { ssum[threadIdx.x] = 0.f; ssq[threadIdx.x] = 0.f; }
  __syncthreads();
  const int c = threadIdx.x & 63;           // stride is a multiple of 64
  float lsum = 0.f, lsq = 0.f;
  for (int i = blockIdx.x * 256 + threadIdx.x; i < kP * 64; i += gridDim.x * 256) {
    float v = agg[i];
    lsum += v; lsq += v * v;
  }
  atomicAdd(&ssum[c], lsum);
  atomicAdd(&ssq[c], lsq);
  __syncthreads();
  if (threadIdx.x < 64) {
    atomicAdd(&slot[threadIdx.x], ssum[threadIdx.x]);
    atomicAdd(&slot[64 + threadIdx.x], ssq[threadIdx.x]);
  }
}

// ---------------- xg = relu(bn(agg) + x), bf16 [P,64] ----------------
__global__ __launch_bounds__(256) void k_apply_xg(const float* __restrict__ agg,
                                                  const float* __restrict__ slot,
                                                  const float* __restrict__ x,
                                                  __bf16* __restrict__ xg) {
  for (int i = blockIdx.x * 256 + threadIdx.x; i < kP * 8; i += gridDim.x * 256) {
    int c0 = (i & 7) * 8, p = i >> 3;
    int n, t, v; decode_p(p, n, t, v);
    const v4f a0 = *(const v4f*)(agg + (size_t)p * 64 + c0);
    const v4f a1 = *(const v4f*)(agg + (size_t)p * 64 + c0 + 4);
    v8bf o;
#pragma unroll
    for (int e = 0; e < 8; ++e) {
      float av = (e < 4) ? a0[e] : a1[e - 4];
      float g = av * slot[128 + c0 + e] + slot[192 + c0 + e] +
                x[((n * 64 + c0 + e) * kT + t) * kV + v];
      o[e] = (__bf16)fmaxf(g, 0.f);
    }
    *(v8bf*)(xg + (size_t)p * 64 + c0) = o;
  }
}

// ---- dilated temporal conv 16->16, K=5: channel-pair/thread, v_pk_fma ----------
__global__ __launch_bounds__(256) void k_tconv(const __bf16* __restrict__ act,
                                               const float* __restrict__ w,   // [16][16][5]
                                               const float* __restrict__ tb,  // [16]
                                               __bf16* __restrict__ out,
                                               float* __restrict__ stats, int dil) {
  __shared__ v2f sw2[8 * 16 * 5];   // [cp][m][tau] -> (w[2cp], w[2cp+1])
  __shared__ float ssum[16], ssq[16];
  for (int i = threadIdx.x; i < 8 * 16 * 5; i += 256) {
    int tau = i % 5, t2 = i / 5;
    int m = t2 & 15, cp = t2 >> 4, c0 = cp * 2;
    sw2[i] = v2f{w[(c0 * 16 + m) * 5 + tau], w[((c0 + 1) * 16 + m) * 5 + tau]};
  }
  if (threadIdx.x < 16) { ssum[threadIdx.x] = 0.f; ssq[threadIdx.x] = 0.f; }
  __syncthreads();

  const int cp = (blockIdx.x * 256 + threadIdx.x) & 7;  // stride multiple of 8
  const int c0 = cp * 2;
  v2f lsum = {0.f, 0.f}, lsq = {0.f, 0.f};
  for (int i = blockIdx.x * 256 + threadIdx.x; i < kP * 8; i += gridDim.x * 256) {
    int p = i >> 3;
    int n, t, v; decode_p(p, n, t, v);
    v2f acc = {tb[c0], tb[c0 + 1]};
#pragma unroll
    for (int tau = 0; tau < 5; ++tau) {
      int tt = t + dil * (tau - 2);
      if (tt >= 0 && tt < kT) {
        const v8bf* ar = (const v8bf*)(act + (size_t)((n * kT + tt) * kV + v) * 16);
        v8bf r0 = ar[0], r1 = ar[1];
#pragma unroll
        for (int m = 0; m < 8; ++m)
          acc += sw2[(cp * 16 + m) * 5 + tau] * (float)r0[m];
#pragma unroll
        for (int m = 0; m < 8; ++m)
          acc += sw2[(cp * 16 + 8 + m) * 5 + tau] * (float)r1[m];
      }
    }
    v2bf o; o[0] = (__bf16)acc[0]; o[1] = (__bf16)acc[1];
    *(v2bf*)(out + (size_t)p * 16 + c0) = o;
    lsum += acc; lsq += acc * acc;
  }
  atomicAdd(&ssum[c0], lsum[0]);
  atomicAdd(&ssum[c0 + 1], lsum[1]);
  atomicAdd(&ssq[c0], lsq[0]);
  atomicAdd(&ssq[c0 + 1], lsq[1]);
  __syncthreads();
  if (threadIdx.x < 16) {
    atomicAdd(&stats[threadIdx.x], ssum[threadIdx.x]);
    atomicAdd(&stats[16 + threadIdx.x], ssq[threadIdx.x]);
  }
}

// ---------------- max-pool over T, window 3, pad 1: 8 channels/thread ----------
__global__ __launch_bounds__(256) void k_maxpool(const __bf16* __restrict__ act,
                                                 __bf16* __restrict__ out,
                                                 float* __restrict__ stats) {
  __shared__ float ssum[16], ssq[16];
  if (threadIdx.x < 16) { ssum[threadIdx.x] = 0.f; ssq[threadIdx.x] = 0.f; }
  __syncthreads();
  const int mb = ((blockIdx.x * 256 + threadIdx.x) & 1) * 8;  // stride even
  v8f lsum = {}, lsq = {};
  for (int i = blockIdx.x * 256 + threadIdx.x; i < kP * 2; i += gridDim.x * 256) {
    int p = i >> 1;
    int n, t, v; decode_p(p, n, t, v);
    v8f mx;
#pragma unroll
    for (int e = 0; e < 8; ++e) mx[e] = -3.4e38f;
#pragma unroll
    for (int dt = -1; dt <= 1; ++dt) {
      int tt = t + dt;
      if (tt >= 0 && tt < kT) {
        v8bf r = *(const v8bf*)(act + (size_t)((n * kT + tt) * kV + v) * 16 + mb);
#pragma unroll
        for (int e = 0; e < 8; ++e) mx[e] = fmaxf(mx[e], (float)r[e]);
      }
    }
    v8bf o;
#pragma unroll
    for (int e = 0; e < 8; ++e) {
      o[e] = (__bf16)mx[e];
      lsum[e] += mx[e]; lsq[e] += mx[e] * mx[e];
    }
    *(v8bf*)(out + (size_t)p * 16 + mb) = o;
  }
#pragma unroll
  for (int e = 0; e < 8; ++e) {
    atomicAdd(&ssum[mb + e], lsum[e]);
    atomicAdd(&ssq[mb + e], lsq[e]);
  }
  __syncthreads();
  if (threadIdx.x < 16) {
    atomicAdd(&stats[threadIdx.x], ssum[threadIdx.x]);
    atomicAdd(&stats[16 + threadIdx.x], ssq[threadIdx.x]);
  }
}

// ---------------- final: out[n, kb+m, t, v] = relu(bn(src) + xg) ----------------
__global__ __launch_bounds__(256) void k_final16(const __bf16* __restrict__ src,
                                                 const float* __restrict__ slot,
                                                 const __bf16* __restrict__ xg,
                                                 float* __restrict__ out, int kb) {
  for (int i = blockIdx.x * 256 + threadIdx.x; i < kP * 2; i += gridDim.x * 256) {
    int mb = (i & 1) * 8, p = i >> 1;
    v8bf s = *(const v8bf*)(src + (size_t)p * 16 + mb);
    v8bf xr = *(const v8bf*)(xg + (size_t)p * 64 + kb + mb);
    int n, t, v; decode_p(p, n, t, v);
#pragma unroll
    for (int e = 0; e < 8; ++e) {
      float val = (float)s[e] * slot[32 + mb + e] + slot[48 + mb + e] + (float)xr[e];
      out[((n * 64 + kb + mb + e) * kT + t) * kV + v] = fmaxf(val, 0.f);
    }
  }
}

// ============================ launcher ============================
extern "C" void kernel_launch(void* const* d_in, const int* in_sizes, int n_in,
                              void* d_out, int out_size, void* d_ws, size_t ws_size,
                              hipStream_t stream) {
  const float* x    = (const float*)d_in[0];
  const float* PA   = (const float*)d_in[1];
  const float* gdw  = (const float*)d_in[2];
  const float* gdb  = (const float*)d_in[3];
  const float* gdg  = (const float*)d_in[4];
  const float* gdbt = (const float*)d_in[5];
  const float* gsw  = (const float*)d_in[6];
  const float* gsb  = (const float*)d_in[7];
  const float* gsg  = (const float*)d_in[8];
  const float* gsbt = (const float*)d_in[9];
  const float* gbng = (const float*)d_in[10];
  const float* gbnb = (const float*)d_in[11];
  const float* brw  = (const float*)d_in[12];
  const float* brb  = (const float*)d_in[13];
  const float* brg  = (const float*)d_in[14];
  const float* brbt = (const float*)d_in[15];
  const float* tcw  = (const float*)d_in[16];
  const float* tcb  = (const float*)d_in[17];
  const float* tcg  = (const float*)d_in[18];
  const float* tcbt = (const float*)d_in[19];
  const float* mpg  = (const float*)d_in[20];
  const float* mpb  = (const float*)d_in[21];
  float* out = (float*)d_out;

  char* w = (char*)d_ws;
  size_t off = 0;
  auto carve = [&](size_t bytes) -> void* {
    void* p = w + off;
    off = (off + bytes + 255) & ~(size_t)255;
    return p;
  };
  float*  stats = (float*)carve(2048 * sizeof(float));
  __bf16* gdwb  = (__bf16*)carve(3 * 16 * 64 * 2);
  __bf16* gswb  = (__bf16*)carve(15 * 16 * 32 * 2);
  __bf16* brwb  = (__bf16*)carve(4 * 16 * 64 * 2);
  __bf16* xb    = (__bf16*)carve((size_t)kP * 64 * 2);
  __bf16* xd    = (__bf16*)carve((size_t)kP * 16 * 2);
  __bf16* zv    = (__bf16*)carve((size_t)kP * 16 * 2);
  __bf16* pre   = (__bf16*)carve((size_t)kP * 16 * 2);
  __bf16* act   = (__bf16*)carve((size_t)kP * 16 * 2);
  __bf16* tco   = (__bf16*)carve((size_t)kP * 16 * 2);
  float*  agg   = (float*)carve((size_t)kP * 64 * sizeof(float));
  __bf16* xg    = (__bf16*)carve((size_t)kP * 64 * 2);

  const float invP = 1.0f / (float)kP;
  float* aggSlot = stats + 25 * 64;
  auto slot16 = [&](int s) { return stats + s * 64; };

  k_prep<<<32, 256, 0, stream>>>(gdw, gsw, brw, gdwb, gswb, brwb, stats);
  k_zero_f32<<<4096, 256, 0, stream>>>(agg, kP * 64);
  k_pack_x<<<4096, 256, 0, stream>>>(x, xb);

  // ---- 3 GCN layers ----
  for (int i = 0; i < 3; ++i) {
    k_gemm_nk64<<<512, 256, 0, stream>>>(xb, gdwb + i * 1024, gdb + i * 16, pre,
                                         slot16(i));
    k_finalize16<<<1, 16, 0, stream>>>(slot16(i), gdg + i * 16, gdbt + i * 16, invP);
    k_apply16_relu<<<2048, 256, 0, stream>>>(pre, slot16(i), xd);
    for (int j = 0; j < 5; ++j) {
      int ij = i * 5 + j;
      k_vmix<<<kNT * 8 / 256, 256, 0, stream>>>(xd, PA + ij * kV * kV, zv);
      k_gemm_nk16<<<512, 256, 0, stream>>>(zv, gswb + ij * 512, gsb + ij * 16, pre,
                                           slot16(3 + ij));
      k_finalize16<<<1, 16, 0, stream>>>(slot16(3 + ij), gsg + ij * 16,
                                         gsbt + ij * 16, invP);
      k_accum_agg<<<2048, 256, 0, stream>>>(pre, slot16(3 + ij), agg, j);
    }
  }

  // ---- aggregate BN + residual + relu -> xg ----
  k_stats64<<<2048, 256, 0, stream>>>(agg, aggSlot);
  k_finalize64<<<1, 64, 0, stream>>>(aggSlot, gbng, gbnb, invP);
  k_apply_xg<<<4096, 256, 0, stream>>>(agg, aggSlot, x, xg);

  // ---- branches 0,1: conv -> bn/relu -> dilated tconv -> bn -> out ----
  for (int k = 0; k < 2; ++k) {
    k_gemm_nk64<<<512, 256, 0, stream>>>(xg, brwb + k * 1024, brb + k * 16, pre,
                                         slot16(18 + k));
    k_finalize16<<<1, 16, 0, stream>>>(slot16(18 + k), brg + k * 16, brbt + k * 16,
                                       invP);
    k_apply16_relu<<<2048, 256, 0, stream>>>(pre, slot16(18 + k), act);
    k_tconv<<<2048, 256, 0, stream>>>(act, tcw + k * 16 * 16 * 5, tcb + k * 16, tco,
                                      slot16(22 + k), k + 1);
    k_finalize16<<<1, 16, 0, stream>>>(slot16(22 + k), tcg + k * 16, tcbt + k * 16,
                                       invP);
    k_final16<<<2048, 256, 0, stream>>>(tco, slot16(22 + k), xg, out, k * 16);
  }

  // ---- branch 2: conv -> bn/relu -> maxpool(T,3) -> bn -> out ----
  k_gemm_nk64<<<512, 256, 0, stream>>>(xg, brwb + 2 * 1024, brb + 32, pre, slot16(20));
  k_finalize16<<<1, 16, 0, stream>>>(slot16(20), brg + 32, brbt + 32, invP);
  k_apply16_relu<<<2048, 256, 0, stream>>>(pre, slot16(20), act);
  k_maxpool<<<2048, 256, 0, stream>>>(act, tco, slot16(24));
  k_finalize16<<<1, 16, 0, stream>>>(slot16(24), mpg, mpb, invP);
  k_final16<<<2048, 256, 0, stream>>>(tco, slot16(24), xg, out, 32);

  // ---- branch 3: conv -> bn -> out ----
  k_gemm_nk64<<<512, 256, 0, stream>>>(xg, brwb + 3 * 1024, brb + 48, pre, slot16(21));
  k_finalize16<<<1, 16, 0, stream>>>(slot16(21), brg + 48, brbt + 48, invP);
  k_final16<<<2048, 256, 0, stream>>>(pre, slot16(21), xg, out, 48);
}
